// CausalSelfAttention2_44169443672369
// MI455X (gfx1250) — matrix operand order repaired and checked by reading, not compiled, not run on hardware
//
#include <hip/hip_runtime.h>

// ---------------------------------------------------------------------------
// CDNA5 (gfx1250) wave32 WMMA implementation of the grouped causal attention
// block. All GEMM-like math goes through v_wmma_f32_16x16x32_f16; GEMM B tiles
// are staged through LDS with gfx1250 async global->LDS copies (ASYNCcnt).
// ---------------------------------------------------------------------------

typedef __attribute__((ext_vector_type(8)))  _Float16 v8h;
typedef __attribute__((ext_vector_type(16))) _Float16 v16h;
typedef __attribute__((ext_vector_type(8)))  float    v8f;
typedef __attribute__((ext_vector_type(4)))  int      v4i;

#define WMMA_F16(a, b, c) \
  __builtin_amdgcn_wmma_f32_16x16x32_f16(false, (a), false, (b), (short)0, (c), false, false)

#if defined(__has_builtin)
#if __has_builtin(__builtin_amdgcn_global_load_async_to_lds_b128)
#define HAVE_ASYNC_LDS 1
#endif
#endif
#ifndef HAVE_ASYNC_LDS
#define HAVE_ASYNC_LDS 0
#endif

#if HAVE_ASYNC_LDS
#define AS1 __attribute__((address_space(1)))
#define AS3 __attribute__((address_space(3)))
#endif

// Copy 16 bytes global -> LDS. Async (ASYNCcnt) when available.
__device__ __forceinline__ void copy16B_to_lds(const _Float16* g, _Float16* l) {
#if HAVE_ASYNC_LDS
  __builtin_amdgcn_global_load_async_to_lds_b128((AS1 v4i*)g, (AS3 v4i*)l, 0, 0);
#else
  *(v8h*)l = *(const v8h*)g;
#endif
}

__device__ __forceinline__ void wait_async_stage() {
#if HAVE_ASYNC_LDS
#if __has_builtin(__builtin_amdgcn_s_wait_asynccnt)
  __builtin_amdgcn_s_wait_asynccnt(0);
#else
  asm volatile("s_wait_asynccnt 0x0" ::: "memory");
#endif
#endif
}

__device__ __forceinline__ v8f zero_v8f() {
  v8f z;
#pragma unroll
  for (int i = 0; i < 8; ++i) z[i] = 0.0f;
  return z;
}

__device__ __forceinline__ v16h zero_v16h() {
  v16h z;
#pragma unroll
  for (int i = 0; i < 16; ++i) z[i] = (_Float16)0.0f;
  return z;
}

// A-fragment (16x32 f16, this lane = row M, one contiguous 16-half K run).
// p must already include the +half*16 K offset; reads p[0..15].
__device__ __forceinline__ v16h afrag_from(const _Float16* p) {
  v8h lo = *(const v8h*)p;
  v8h hi = *(const v8h*)(p + 8);
  v16h a;
#pragma unroll
  for (int i = 0; i < 8; ++i) { a[i] = lo[i]; a[i + 8] = hi[i]; }
  return a;
}

// ---------------------------------------------------------------------------
// Elementwise conversion kernels
// ---------------------------------------------------------------------------
__global__ void cvt_f32_to_f16_kernel(const float* __restrict__ src,
                                      _Float16* __restrict__ dst, int n) {
  int i = blockIdx.x * blockDim.x + threadIdx.x;
  if (i < n) dst[i] = (_Float16)src[i];
}

// Wt[n*K + k] = (f16) W[k*N + n]
__global__ void transpose_to_f16_kernel(const float* __restrict__ W,
                                        _Float16* __restrict__ Wt, int K, int N) {
  int i = blockIdx.x * blockDim.x + threadIdx.x;
  if (i < K * N) {
    int n = i / K;
    int k = i - n * K;
    Wt[i] = (_Float16)W[(size_t)k * N + n];
  }
}

// ---------------------------------------------------------------------------
// WMMA GEMM: C[M,N] = A[M,K] * Bt[N,K]^T.
// Block = 256 threads (8 waves), block tile 256x64 (per-wave 32x64).
// The 64x32 B tile is staged in LDS (double buffered) via async global->LDS
// copies and shared by all 8 waves; A fragments stream straight from global.
// MODE 0: silu(C) -> f16 O16a (ld = N)
// MODE 1: f16 out, column-split at nsplit: cols [0,nsplit) -> O16a,
//         [nsplit,N) -> O16b (both ld = nsplit)
// MODE 2: f32 out -> O32 (ld = N)
// ---------------------------------------------------------------------------
template <int MODE>
__global__ __launch_bounds__(256) void wmma_gemm_kernel(
    const _Float16* __restrict__ A, const _Float16* __restrict__ Bt,
    _Float16* __restrict__ O16a, _Float16* __restrict__ O16b,
    float* __restrict__ O32, int N, int K, int nsplit) {
  __shared__ alignas(32) _Float16 bsh[2][64 * 32];  // [stage][n_local*32 + k_local]

  const int tid  = threadIdx.x;
  const int wave = tid >> 5;
  const int lane = tid & 31;
  const int half = lane >> 4;
  const int l16  = lane & 15;

  const int n0 = blockIdx.x * 64;
  const int m0 = blockIdx.y * 256 + wave * 32;

  // B-tile copy assignment: 64 rows x 32 halfs; one 16B segment per thread.
  const int brow = tid >> 2;   // 0..63
  const int bseg = tid & 3;    // 0..3
  const _Float16* bsrc = Bt + (size_t)(n0 + brow) * K + bseg * 8;
  _Float16* bdst[2] = { &bsh[0][brow * 32 + bseg * 8], &bsh[1][brow * 32 + bseg * 8] };

  v8f acc[2][4];
#pragma unroll
  for (int mi = 0; mi < 2; ++mi)
#pragma unroll
    for (int t = 0; t < 4; ++t) acc[mi][t] = zero_v8f();

  const _Float16* arow0 = A + (size_t)(m0 + l16) * K;
  const _Float16* arow1 = arow0 + (size_t)16 * K;

  const int nsteps = K / 32;

  // Prologue: stage 0 into buffer 0.
  copy16B_to_lds(bsrc, bdst[0]);

  for (int s = 0; s < nsteps; ++s) {
    const int k0 = s * 32;

    wait_async_stage();   // this wave's async copies done (ASYNCcnt)
    __syncthreads();      // all waves' copies of current stage visible

    if (s + 1 < nsteps)   // prefetch next stage into the other buffer
      copy16B_to_lds(bsrc + k0 + 32, bdst[(s + 1) & 1]);

    const _Float16* bstage = bsh[s & 1];
    v16h a0 = afrag_from(arow0 + k0 + half * 16);
    v16h a1 = afrag_from(arow1 + k0 + half * 16);
#pragma unroll
    for (int t = 0; t < 4; ++t) {
      v16h b = *(const v16h*)&bstage[(t * 16 + l16) * 32 + half * 16];
      acc[0][t] = WMMA_F16(a0, b, acc[0][t]);
      acc[1][t] = WMMA_F16(a1, b, acc[1][t]);
    }
  }

#pragma unroll
  for (int mi = 0; mi < 2; ++mi) {
#pragma unroll
    for (int t = 0; t < 4; ++t) {
#pragma unroll
      for (int r = 0; r < 8; ++r) {
        const int row = m0 + mi * 16 + half * 8 + r;
        const int col = n0 + t * 16 + l16;
        float v = acc[mi][t][r];
        if (MODE == 0) {
          v = v / (1.0f + __expf(-v));  // silu
          O16a[(size_t)row * N + col] = (_Float16)v;
        } else if (MODE == 1) {
          if (col < nsplit)
            O16a[(size_t)row * nsplit + col] = (_Float16)v;
          else
            O16b[(size_t)row * nsplit + (col - nsplit)] = (_Float16)v;
        } else {
          O32[(size_t)row * N + col] = v;
        }
      }
    }
  }
}

// ---------------------------------------------------------------------------
// Per-(b,h,g) column means of q,k,v head blocks -> synthetic tokens (f16).
// grid: 512 blocks (= (b*16+h)*4+g), 64 threads (= d)
// ---------------------------------------------------------------------------
__global__ void means_kernel(const _Float16* __restrict__ Qh,
                             const _Float16* __restrict__ Kh,
                             const _Float16* __restrict__ Vh,
                             _Float16* __restrict__ SQ,
                             _Float16* __restrict__ SK,
                             _Float16* __restrict__ SV) {
  const int d   = threadIdx.x;
  const int idx = blockIdx.x;
  const int g = idx & 3;
  const int h = (idx >> 2) & 15;
  const int b = idx >> 6;
  const size_t base = ((size_t)(b * 2048 + g * 512)) * 1024 + h * 64 + d;
  float sq = 0.f, sk = 0.f, sv = 0.f;
  for (int t = 0; t < 512; ++t) {
    const size_t off = base + (size_t)t * 1024;
    sq += (float)Qh[off];
    sk += (float)Kh[off];
    sv += (float)Vh[off];
  }
  const float inv = 1.0f / 512.0f;
  SQ[idx * 64 + d] = (_Float16)(sq * inv);
  SK[idx * 64 + d] = (_Float16)(sk * inv);
  SV[idx * 64 + d] = (_Float16)(sv * inv);
}

// ---------------------------------------------------------------------------
// Flash attention per (b,h,g): S = 513 (512 tokens + synthetic mean token),
// d = 64. One wave per 16-query tile. grid = (33, h*4+g, b), block = 32.
// Scores: 2x WMMA (K=32 halves of d=64). PV: P restaged via LDS as A-frag,
// V chunk transposed into LDS for contiguous B-frags.
// ---------------------------------------------------------------------------
__global__ void attn_kernel(const _Float16* __restrict__ Qh,
                            const _Float16* __restrict__ Kh,
                            const _Float16* __restrict__ Vh,
                            const _Float16* __restrict__ SQ,
                            const _Float16* __restrict__ SK,
                            const _Float16* __restrict__ SV,
                            _Float16* __restrict__ XO,
                            float* __restrict__ OSYN) {
  __shared__ alignas(32) _Float16 p_lds[16 * 32];   // P tile [m][key]
  __shared__ alignas(32) _Float16 vt_lds[64 * 32];  // V^T    [d][key]

  const int lane = threadIdx.x & 31;
  const int half = lane >> 4;
  const int l16  = lane & 15;

  const int qt = blockIdx.x;        // 0..32 (32 = synthetic-query tile)
  const int h  = blockIdx.y >> 2;
  const int g  = blockIdx.y & 3;
  const int b  = blockIdx.z;

  const int rowbase = b * 2048 + g * 512;  // token-row base in q/k/v
  const int col0    = h * 64;
  const int sidx    = (b * 16 + h) * 4 + g;
  const bool synq   = (qt == 32);
  const int q0      = qt * 16;

  // ---- Q fragments (held for the whole block) ----
  v16h aq0, aq1;
  if (!synq) {
    const _Float16* qp = Qh + (size_t)(rowbase + q0 + l16) * 1024 + col0 + half * 16;
    aq0 = afrag_from(qp);
    aq1 = afrag_from(qp + 32);
  } else {
    aq0 = zero_v16h();
    aq1 = zero_v16h();
    if (l16 == 0) {  // only query 512 (the mean token) is real
      const _Float16* qp = SQ + (size_t)sidx * 64 + half * 16;
      aq0 = afrag_from(qp);
      aq1 = afrag_from(qp + 32);
    }
  }

  float rmax[8], rsum[8];
#pragma unroll
  for (int r = 0; r < 8; ++r) { rmax[r] = -3.0e38f; rsum[r] = 0.0f; }
  v8f o[4];
#pragma unroll
  for (int t = 0; t < 4; ++t) o[t] = zero_v8f();

  const int nch = synq ? 17 : ((q0 + 16 + 31) >> 5);  // 32-key chunks

  for (int c = 0; c < nch; ++c) {
    const int kb = c * 32;

    // ---- scores: two 16-key tiles ----
    v8f s[2];
#pragma unroll
    for (int sct = 0; sct < 2; ++sct) {
      const int key = kb + sct * 16 + l16;
      v16h bk0, bk1;
      if (key < 512) {
        const _Float16* kp = Kh + (size_t)(rowbase + key) * 1024 + col0 + half * 16;
        bk0 = *(const v16h*)kp;
        bk1 = *(const v16h*)(kp + 32);
      } else if (key == 512) {
        const _Float16* kp = SK + (size_t)sidx * 64 + half * 16;
        bk0 = *(const v16h*)kp;
        bk1 = *(const v16h*)(kp + 32);
      } else {
        bk0 = zero_v16h();
        bk1 = zero_v16h();
      }
      v8f st = zero_v8f();
      st = WMMA_F16(aq0, bk0, st);
      st = WMMA_F16(aq1, bk1, st);
      s[sct] = st;
    }

    // ---- scale + causal mask + online softmax ----
    float alpha[8];
#pragma unroll
    for (int r = 0; r < 8; ++r) {
      const int m  = half * 8 + r;
      const int qi = q0 + m;
      float a0v = s[0][r] * 0.125f;
      float a1v = s[1][r] * 0.125f;
      const int key0 = kb + l16;
      const int key1 = kb + 16 + l16;
      if (key0 > qi || key0 > 512 || qi > 512) a0v = -3.0e38f;
      if (key1 > qi || key1 > 512 || qi > 512) a1v = -3.0e38f;

      float mx = fmaxf(a0v, a1v);
      mx = fmaxf(mx, __shfl_xor(mx, 1));
      mx = fmaxf(mx, __shfl_xor(mx, 2));
      mx = fmaxf(mx, __shfl_xor(mx, 4));
      mx = fmaxf(mx, __shfl_xor(mx, 8));

      const float nm = fmaxf(rmax[r], mx);
      const float al = __expf(rmax[r] - nm);
      rmax[r] = nm;

      const float p0 = __expf(a0v - nm);
      const float p1 = __expf(a1v - nm);
      float ps = p0 + p1;
      ps += __shfl_xor(ps, 1);
      ps += __shfl_xor(ps, 2);
      ps += __shfl_xor(ps, 4);
      ps += __shfl_xor(ps, 8);
      rsum[r] = rsum[r] * al + ps;
      alpha[r] = al;

      p_lds[m * 32 + l16]      = (_Float16)p0;
      p_lds[m * 32 + 16 + l16] = (_Float16)p1;
    }

    // ---- rescale running output ----
#pragma unroll
    for (int t = 0; t < 4; ++t)
#pragma unroll
      for (int r = 0; r < 8; ++r) o[t][r] *= alpha[r];

    // ---- stage V^T chunk: vt_lds[d][key-kb], one key-row per lane ----
    {
      const int key = kb + lane;
      if (key < 512) {
        const v8h* vp = (const v8h*)(Vh + (size_t)(rowbase + key) * 1024 + col0);
#pragma unroll
        for (int c8 = 0; c8 < 8; ++c8) {
          v8h vv = vp[c8];
#pragma unroll
          for (int i = 0; i < 8; ++i) vt_lds[(c8 * 8 + i) * 32 + lane] = vv[i];
        }
      } else if (key == 512) {
        const v8h* vp = (const v8h*)(SV + (size_t)sidx * 64);
#pragma unroll
        for (int c8 = 0; c8 < 8; ++c8) {
          v8h vv = vp[c8];
#pragma unroll
          for (int i = 0; i < 8; ++i) vt_lds[(c8 * 8 + i) * 32 + lane] = vv[i];
        }
      } else {
#pragma unroll
        for (int d = 0; d < 64; ++d) vt_lds[d * 32 + lane] = (_Float16)0.0f;
      }
    }

    // ---- P A-frag from LDS (same-wave LDS ops are in order) ----
    v16h ap;
    {
      const _Float16* pp = &p_lds[l16 * 32 + half * 16];
      ap = afrag_from(pp);
    }

    // ---- PV accumulation ----
#pragma unroll
    for (int t = 0; t < 4; ++t) {
      const v16h bv = *(const v16h*)&vt_lds[(t * 16 + l16) * 32 + half * 16];
      o[t] = WMMA_F16(ap, bv, o[t]);
    }
  }

  // ---- normalize + store ----
  float inv[8];
#pragma unroll
  for (int r = 0; r < 8; ++r) inv[r] = 1.0f / rsum[r];
#pragma unroll
  for (int t = 0; t < 4; ++t) {
#pragma unroll
    for (int r = 0; r < 8; ++r) {
      const int m   = half * 8 + r;
      const int tok = q0 + m;
      const float val = o[t][r] * inv[r];
      const int d = t * 16 + l16;
      if (tok < 512) {
        // xo layout: [b*2048 + g*512 + tok][h*64 + d]  (feeds final GEMM)
        XO[(size_t)(rowbase + tok) * 1024 + col0 + d] = (_Float16)val;
      } else if (tok == 512) {
        OSYN[(size_t)sidx * 64 + d] = val;  // synthetic-row attention output
      }
    }
  }
}

// ---------------------------------------------------------------------------
// Tiny 3-token causal attention per (b,h) over the group synthetic tokens,
// plus writes y = (qg, kg, vg). grid = 128 (= b*16+h), block = 32 (2 d/lane).
// ---------------------------------------------------------------------------
__global__ void group_attn_kernel(const _Float16* __restrict__ SQ,
                                  const _Float16* __restrict__ SK,
                                  const float* __restrict__ OSYN,
                                  float* __restrict__ YQ,
                                  float* __restrict__ YK,
                                  float* __restrict__ YV) {
  const int lane = threadIdx.x;  // 0..31
  const int idx  = blockIdx.x;   // b*16 + h
  float q[3][2], k[3][2], vv[3][2];
#pragma unroll
  for (int g = 0; g < 3; ++g) {
    const int sb = (idx * 4 + g) * 64 + lane * 2;
    q[g][0]  = (float)SQ[sb];   q[g][1]  = (float)SQ[sb + 1];
    k[g][0]  = (float)SK[sb];   k[g][1]  = (float)SK[sb + 1];
    vv[g][0] = OSYN[sb];        vv[g][1] = OSYN[sb + 1];
  }
  float s[3][3];
#pragma unroll
  for (int i = 0; i < 3; ++i) {
#pragma unroll
    for (int j = 0; j < 3; ++j) {
      if (j <= i) {
        float p = q[i][0] * k[j][0] + q[i][1] * k[j][1];
        p += __shfl_xor(p, 1);
        p += __shfl_xor(p, 2);
        p += __shfl_xor(p, 4);
        p += __shfl_xor(p, 8);
        p += __shfl_xor(p, 16);
        s[i][j] = p * 0.125f;
      } else {
        s[i][j] = -3.0e38f;
      }
    }
  }
#pragma unroll
  for (int i = 0; i < 3; ++i) {
    float mx = -3.0e38f;
#pragma unroll
    for (int j = 0; j <= i; ++j) mx = fmaxf(mx, s[i][j]);
    float p[3], den = 0.0f;
#pragma unroll
    for (int j = 0; j <= i; ++j) { p[j] = __expf(s[i][j] - mx); den += p[j]; }
    float o0 = 0.0f, o1 = 0.0f;
#pragma unroll
    for (int j = 0; j <= i; ++j) { o0 += p[j] * vv[j][0]; o1 += p[j] * vv[j][1]; }
    const float id = 1.0f / den;
    const int ob = (idx * 3 + i) * 64 + lane * 2;
    YV[ob]     = o0 * id;
    YV[ob + 1] = o1 * id;
    YQ[ob]     = q[i][0];
    YQ[ob + 1] = q[i][1];
    YK[ob]     = k[i][0];
    YK[ob + 1] = k[i][1];
  }
}

// ---------------------------------------------------------------------------
// Host-side launcher
// ---------------------------------------------------------------------------
extern "C" void kernel_launch(void* const* d_in, const int* in_sizes, int n_in,
                              void* d_out, int out_size, void* d_ws, size_t ws_size,
                              hipStream_t stream) {
  const int B = 8, T = 2048, DIM = 1024;
  const int M = B * T;  // 16384

  const float* x   = (const float*)d_in[0];
  const float* Wv  = (const float*)d_in[1];
  const float* Wqk = (const float*)d_in[2];
  const float* Wo  = (const float*)d_in[3];
  float* out = (float*)d_out;

  char* ws = (char*)d_ws;
  size_t off = 0;
  auto alloc = [&](size_t bytes) -> void* {
    void* p = ws + off;
    off += (bytes + 255) & ~(size_t)255;
    return p;
  };

  _Float16* x_h   = (_Float16*)alloc((size_t)M * DIM * 2);
  _Float16* wv_t  = (_Float16*)alloc((size_t)DIM * DIM * 2);
  _Float16* wqk_t = (_Float16*)alloc((size_t)2 * DIM * DIM * 2);
  _Float16* wo_t  = (_Float16*)alloc((size_t)DIM * DIM * 2);
  _Float16* v_h   = (_Float16*)alloc((size_t)M * DIM * 2);
  _Float16* q_h   = (_Float16*)alloc((size_t)M * DIM * 2);
  _Float16* k_h   = (_Float16*)alloc((size_t)M * DIM * 2);
  _Float16* xo_h  = (_Float16*)alloc((size_t)M * DIM * 2);
  _Float16* sq    = (_Float16*)alloc((size_t)512 * 64 * 2);
  _Float16* sk    = (_Float16*)alloc((size_t)512 * 64 * 2);
  _Float16* sv    = (_Float16*)alloc((size_t)512 * 64 * 2);
  float*    osyn  = (float*)alloc((size_t)512 * 64 * 4);

  // 1) conversions
  {
    int n = M * DIM;
    cvt_f32_to_f16_kernel<<<(n + 255) / 256, 256, 0, stream>>>(x, x_h, n);
    int nw = DIM * DIM;
    transpose_to_f16_kernel<<<(nw + 255) / 256, 256, 0, stream>>>(Wv, wv_t, DIM, DIM);
    int nqk = DIM * 2 * DIM;
    transpose_to_f16_kernel<<<(nqk + 255) / 256, 256, 0, stream>>>(Wqk, wqk_t, DIM, 2 * DIM);
    transpose_to_f16_kernel<<<(nw + 255) / 256, 256, 0, stream>>>(Wo, wo_t, DIM, DIM);
  }

  // 2) v = silu(x @ Wv)
  wmma_gemm_kernel<0><<<dim3(DIM / 64, M / 256), 256, 0, stream>>>(
      x_h, wv_t, v_h, nullptr, nullptr, DIM, DIM, DIM);

  // 3) qk = v @ Wqk  -> q_h | k_h
  wmma_gemm_kernel<1><<<dim3(2 * DIM / 64, M / 256), 256, 0, stream>>>(
      v_h, wqk_t, q_h, k_h, nullptr, 2 * DIM, DIM, DIM);

  // 4) synthetic mean tokens per (b,h,g)
  means_kernel<<<512, 64, 0, stream>>>(q_h, k_h, v_h, sq, sk, sv);

  // 5) grouped causal attention (S=513, d=64), flash style
  attn_kernel<<<dim3(33, 64, B), 32, 0, stream>>>(q_h, k_h, v_h, sq, sk, sv, xo_h, osyn);

  // 6) tiny group attention + y outputs (appended after `out` in d_out)
  {
    float* yq = out + (size_t)M * DIM;
    float* yk = yq + (size_t)B * 16 * 3 * 64;
    float* yv = yk + (size_t)B * 16 * 3 * 64;
    group_attn_kernel<<<128, 32, 0, stream>>>(sq, sk, osyn, yq, yk, yv);
  }

  // 7) out = xo @ Wo
  wmma_gemm_kernel<2><<<dim3(DIM / 64, M / 256), 256, 0, stream>>>(
      xo_h, wo_t, nullptr, nullptr, out, DIM, DIM, DIM);
}
